// DotProductAttention_51582557225635
// MI455X (gfx1250) — matrix operand order repaired and checked by reading, not compiled, not run on hardware
//
#include <hip/hip_runtime.h>

// Problem constants (match reference).
constexpr int N_  = 8;
constexpr int TD_ = 1024;
constexpr int TE_ = 2048;
constexpr int D_  = 1024;

// LDS column stride (elements) for staged B tiles: 40*2B = 80B = 20 banks ->
// 16 half-wave lanes land 4 banks apart (conflict-free b128 reads), 16B aligned.
constexpr int BSTRIDE = 40;

typedef __attribute__((ext_vector_type(16))) __bf16 bf16x16;
typedef __attribute__((ext_vector_type(8)))  __bf16 bf16x8;
typedef __attribute__((ext_vector_type(4)))  __bf16 bf16x4;
typedef __attribute__((ext_vector_type(8)))  float  f32x8;
typedef __attribute__((ext_vector_type(4)))  float  f32x4;

struct HL { __bf16 h; __bf16 l; };

static __device__ __forceinline__ HL splitf(float x) {
    HL r;
    r.h = (__bf16)x;                    // RNE to bf16 (hi)
    r.l = (__bf16)(x - (float)r.h);     // residual (lo)
    return r;
}

static __device__ __forceinline__ bf16x16 cat8(bf16x8 a, bf16x8 b) {
    bf16x16 r;
#pragma unroll
    for (int i = 0; i < 8; ++i) { r[i] = a[i]; r[i + 8] = b[i]; }
    return r;
}

// ---------------------------------------------------------------------------
// Kernel 1: elementwise fp32 -> (bf16 hi, bf16 lo) split (for Q).
// ---------------------------------------------------------------------------
__global__ __launch_bounds__(256) void dpa_split_kernel(
    const float* __restrict__ x, __bf16* __restrict__ hi, __bf16* __restrict__ lo) {
    size_t i = ((size_t)blockIdx.x * 256 + threadIdx.x) * 4;
    f32x4 v = *(const f32x4*)(x + i);
    bf16x4 h, l;
#pragma unroll
    for (int j = 0; j < 4; ++j) {
        HL s = splitf(v[j]);
        h[j] = s.h; l[j] = s.l;
    }
    *(bf16x4*)(hi + i) = h;
    *(bf16x4*)(lo + i) = l;
}

// ---------------------------------------------------------------------------
// Kernel 2: split M (row-major hi/lo) AND produce transposed M^T hi/lo
// (layout [n, d, s]) via a 32x32 LDS-tiled transpose.
// ---------------------------------------------------------------------------
__global__ __launch_bounds__(256) void dpa_split_transpose_m_kernel(
    const float* __restrict__ M,
    __bf16* __restrict__ Mh,  __bf16* __restrict__ Ml,
    __bf16* __restrict__ MTh, __bf16* __restrict__ MTl) {
    __shared__ float tile[32][33];
    const int n  = blockIdx.z;
    const int bx = blockIdx.x * 32;   // d
    const int by = blockIdx.y * 32;   // s
    const int tx = threadIdx.x & 31;
    const int ry = threadIdx.x >> 5;  // 0..7
#pragma unroll
    for (int i = 0; i < 4; ++i) {
        int r = ry + i * 8;
        int s = by + r, d = bx + tx;
        size_t idx = ((size_t)n * TE_ + s) * D_ + d;
        float v = M[idx];
        HL hl = splitf(v);
        Mh[idx] = hl.h;
        Ml[idx] = hl.l;
        tile[r][tx] = v;
    }
    __syncthreads();
#pragma unroll
    for (int i = 0; i < 4; ++i) {
        int r = ry + i * 8;
        int d = bx + r, s = by + tx;
        float v = tile[tx][r];
        HL hl = splitf(v);
        size_t idx = ((size_t)n * D_ + d) * TE_ + s;
        MTh[idx] = hl.h;
        MTl[idx] = hl.l;
    }
}

// ---------------------------------------------------------------------------
// Kernel 3: S = (Q M^T) / 32, masked with -inf for s >= lengths[n].
// 8 waves/block; block tile 128x128; wave owns a 16x128 strip.
// B chunk (shared by all waves) is staged through LDS, double-buffered,
// one __syncthreads per K-chunk. bf16x3: acc += Ah*Bh + Ah*Bl + Al*Bh.
// ---------------------------------------------------------------------------
__global__ __launch_bounds__(256) void dpa_qk_kernel(
    const __bf16* __restrict__ Qh, const __bf16* __restrict__ Ql,
    const __bf16* __restrict__ Mh, const __bf16* __restrict__ Ml,
    const int* __restrict__ lengths, float* __restrict__ S) {
    __shared__ alignas(16) __bf16 sB[2][2][128 * BSTRIDE];   // [buf][hi/lo][col][K]

    const int n    = blockIdx.z;
    const int cblk = blockIdx.x * 128;                 // s tile
    const int wave = threadIdx.x >> 5;
    const int lane = threadIdx.x & 31;
    const int r0   = blockIdx.y * 128 + wave * 16;     // t tile
    const int lrow = lane & 15;
    const int half = lane >> 4;
    const int koffA = half * 8;    // A frag: K = k0+koffA..+7 and k0+16+koffA..+7
    const int koffB = half * 8;    // B frag: K = k0+koffB..+7 and k0+16+koffB..+7

    const __bf16* qh = Qh + ((size_t)n * TD_ + r0 + lrow) * D_;
    const __bf16* ql = Ql + ((size_t)n * TD_ + r0 + lrow) * D_;

    // Cooperative B staging: thread -> (col, hi/lo part), 64B per chunk each.
    const int tcol  = threadIdx.x & 127;
    const int tpart = threadIdx.x >> 7;
    const __bf16* gB = (tpart ? Ml : Mh) + ((size_t)n * TE_ + cblk + tcol) * D_;
    __bf16* sBme0 = &sB[0][tpart][tcol * BSTRIDE];
    __bf16* sBme1 = &sB[1][tpart][tcol * BSTRIDE];

    const f32x8 zero = {0.f, 0.f, 0.f, 0.f, 0.f, 0.f, 0.f, 0.f};
    f32x8 acc[8];
#pragma unroll
    for (int st = 0; st < 8; ++st) acc[st] = zero;

    bf16x8 rB[4];
#pragma unroll
    for (int i = 0; i < 4; ++i) rB[i] = *(const bf16x8*)(gB + i * 8);

    for (int k0 = 0; k0 < D_; k0 += 32) {
        __bf16* sme = ((k0 >> 5) & 1) ? sBme1 : sBme0;
        const __bf16* sh = &sB[(k0 >> 5) & 1][0][0];
        const __bf16* sl = &sB[(k0 >> 5) & 1][1][0];
#pragma unroll
        for (int i = 0; i < 4; ++i) *(bf16x8*)(sme + i * 8) = rB[i];
        if (k0 + 32 < D_) {
#pragma unroll
            for (int i = 0; i < 4; ++i) rB[i] = *(const bf16x8*)(gB + k0 + 32 + i * 8);
        }
        bf16x16 Ah = cat8(*(const bf16x8*)(qh + k0 + koffA),
                          *(const bf16x8*)(qh + k0 + 16 + koffA));
        bf16x16 Al = cat8(*(const bf16x8*)(ql + k0 + koffA),
                          *(const bf16x8*)(ql + k0 + 16 + koffA));
        __syncthreads();
#pragma unroll
        for (int st = 0; st < 8; ++st) {
            const __bf16* bh = sh + (st * 16 + lrow) * BSTRIDE + koffB;
            const __bf16* bl = sl + (st * 16 + lrow) * BSTRIDE + koffB;
            bf16x16 Bh = cat8(*(const bf16x8*)bh, *(const bf16x8*)(bh + 16));
            bf16x16 Bl = cat8(*(const bf16x8*)bl, *(const bf16x8*)(bl + 16));
            acc[st] = __builtin_amdgcn_wmma_f32_16x16x32_bf16(false, Ah, false, Bh, (short)0, acc[st], false, false);
            acc[st] = __builtin_amdgcn_wmma_f32_16x16x32_bf16(false, Ah, false, Bl, (short)0, acc[st], false, false);
            acc[st] = __builtin_amdgcn_wmma_f32_16x16x32_bf16(false, Al, false, Bh, (short)0, acc[st], false, false);
        }
    }

    const int len = lengths[n];
#pragma unroll
    for (int st = 0; st < 8; ++st) {
        int col = cblk + st * 16 + lrow;
        bool oob = col >= len;
#pragma unroll
        for (int j = 0; j < 8; ++j) {
            int row = r0 + j + half * 8;   // C/D layout: VGPR j -> M = j (+8 for hi lanes)
            float v = oob ? -__builtin_inff() : acc[st][j] * 0.03125f;  // 1/sqrt(1024)
            S[((size_t)n * TD_ + row) * TE_ + col] = v;
        }
    }
}

// ---------------------------------------------------------------------------
// Kernel 4: in-place row softmax over TE=2048. One block per row; values
// kept in registers, LDS tree reductions for max and sum. exp(-inf)=0.
// ---------------------------------------------------------------------------
__global__ __launch_bounds__(256) void dpa_softmax_kernel(float* __restrict__ S) {
    __shared__ float red[256];
    float* p = S + (size_t)blockIdx.x * TE_;
    const int t = threadIdx.x;

    float v[8];
    float m = -__builtin_inff();
#pragma unroll
    for (int i = 0; i < 8; ++i) {
        v[i] = p[t + i * 256];
        m = fmaxf(m, v[i]);
    }
    red[t] = m;
    __syncthreads();
    for (int s = 128; s > 0; s >>= 1) {
        if (t < s) red[t] = fmaxf(red[t], red[t + s]);
        __syncthreads();
    }
    m = red[0];
    __syncthreads();

    float sum = 0.f;
#pragma unroll
    for (int i = 0; i < 8; ++i) {
        float e = (v[i] == -__builtin_inff()) ? 0.f : __expf(v[i] - m);
        v[i] = e;
        sum += e;
    }
    red[t] = sum;
    __syncthreads();
    for (int s = 128; s > 0; s >>= 1) {
        if (t < s) red[t] += red[t + s];
        __syncthreads();
    }
    float inv = 1.f / red[0];
#pragma unroll
    for (int i = 0; i < 8; ++i) p[t + i * 256] = v[i] * inv;
}

// ---------------------------------------------------------------------------
// Kernel 5: contexts = A * M. A (fp32 alignments) hi/lo-split on the fly in
// registers; B from pre-transposed M^T hi/lo, staged through LDS like GEMM1.
// ---------------------------------------------------------------------------
__global__ __launch_bounds__(256) void dpa_av_kernel(
    const float* __restrict__ A,
    const __bf16* __restrict__ MTh, const __bf16* __restrict__ MTl,
    float* __restrict__ C) {
    __shared__ alignas(16) __bf16 sB[2][2][128 * BSTRIDE];   // [buf][hi/lo][col][K]

    const int n    = blockIdx.z;
    const int cblk = blockIdx.x * 128;                 // d tile
    const int wave = threadIdx.x >> 5;
    const int lane = threadIdx.x & 31;
    const int r0   = blockIdx.y * 128 + wave * 16;     // t tile
    const int lrow = lane & 15;
    const int half = lane >> 4;
    const int koffA = half * 8;
    const int koffB = half * 8;

    const float* ar = A + ((size_t)n * TD_ + r0 + lrow) * TE_;

    const int tcol  = threadIdx.x & 127;
    const int tpart = threadIdx.x >> 7;
    const __bf16* gB = (tpart ? MTl : MTh) + ((size_t)n * D_ + cblk + tcol) * TE_;
    __bf16* sBme0 = &sB[0][tpart][tcol * BSTRIDE];
    __bf16* sBme1 = &sB[1][tpart][tcol * BSTRIDE];

    const f32x8 zero = {0.f, 0.f, 0.f, 0.f, 0.f, 0.f, 0.f, 0.f};
    f32x8 acc[8];
#pragma unroll
    for (int st = 0; st < 8; ++st) acc[st] = zero;

    bf16x8 rB[4];
#pragma unroll
    for (int i = 0; i < 4; ++i) rB[i] = *(const bf16x8*)(gB + i * 8);

    for (int k0 = 0; k0 < TE_; k0 += 32) {
        __bf16* sme = ((k0 >> 5) & 1) ? sBme1 : sBme0;
        const __bf16* sh = &sB[(k0 >> 5) & 1][0][0];
        const __bf16* sl = &sB[(k0 >> 5) & 1][1][0];
#pragma unroll
        for (int i = 0; i < 4; ++i) *(bf16x8*)(sme + i * 8) = rB[i];
        if (k0 + 32 < TE_) {
#pragma unroll
            for (int i = 0; i < 4; ++i) rB[i] = *(const bf16x8*)(gB + k0 + 32 + i * 8);
        }
        // Build compensated A fragments from fp32 alignments.
        const float* a0 = ar + k0 + koffA;
        const float* a1 = ar + k0 + 16 + koffA;
        f32x4 p0 = *(const f32x4*)a0;
        f32x4 p1 = *(const f32x4*)(a0 + 4);
        f32x4 p2 = *(const f32x4*)a1;
        f32x4 p3 = *(const f32x4*)(a1 + 4);
        bf16x16 Ah, Al;
#pragma unroll
        for (int i = 0; i < 4; ++i) {
            HL s0 = splitf(p0[i]); Ah[i]      = s0.h; Al[i]      = s0.l;
            HL s1 = splitf(p1[i]); Ah[4 + i]  = s1.h; Al[4 + i]  = s1.l;
            HL s2 = splitf(p2[i]); Ah[8 + i]  = s2.h; Al[8 + i]  = s2.l;
            HL s3 = splitf(p3[i]); Ah[12 + i] = s3.h; Al[12 + i] = s3.l;
        }
        __syncthreads();
#pragma unroll
        for (int st = 0; st < 8; ++st) {
            const __bf16* bh = sh + (st * 16 + lrow) * BSTRIDE + koffB;
            const __bf16* bl = sl + (st * 16 + lrow) * BSTRIDE + koffB;
            bf16x16 Bh = cat8(*(const bf16x8*)bh, *(const bf16x8*)(bh + 16));
            bf16x16 Bl = cat8(*(const bf16x8*)bl, *(const bf16x8*)(bl + 16));
            acc[st] = __builtin_amdgcn_wmma_f32_16x16x32_bf16(false, Ah, false, Bh, (short)0, acc[st], false, false);
            acc[st] = __builtin_amdgcn_wmma_f32_16x16x32_bf16(false, Ah, false, Bl, (short)0, acc[st], false, false);
            acc[st] = __builtin_amdgcn_wmma_f32_16x16x32_bf16(false, Al, false, Bh, (short)0, acc[st], false, false);
        }
    }

#pragma unroll
    for (int st = 0; st < 8; ++st) {
        int col = cblk + st * 16 + lrow;
#pragma unroll
        for (int j = 0; j < 8; ++j) {
            int row = r0 + j + half * 8;
            C[((size_t)n * TD_ + row) * D_ + col] = acc[st][j];
        }
    }
}

// ---------------------------------------------------------------------------
// Launch: split Q -> split+transpose M -> QK^T+mask -> softmax -> A*M.
// Workspace layout (elements): Qh|Ql (8M each), Mh|Ml (16M each),
// MTh|MTl (16M each) => 160 MB total, all bf16 (fits MI455X 192MB L2).
// ---------------------------------------------------------------------------
extern "C" void kernel_launch(void* const* d_in, const int* in_sizes, int n_in,
                              void* d_out, int out_size, void* d_ws, size_t ws_size,
                              hipStream_t stream) {
    (void)in_sizes; (void)n_in; (void)out_size; (void)ws_size;

    const float* Q       = (const float*)d_in[0];
    const float* M       = (const float*)d_in[1];
    const int*   lengths = (const int*)d_in[2];

    float* ctx   = (float*)d_out;                       // [N, TD, D]
    float* align = ctx + (size_t)N_ * TD_ * D_;         // [N, TD, TE]

    const size_t qsz = (size_t)N_ * TD_ * D_;           // 8M elements
    const size_t msz = (size_t)N_ * TE_ * D_;           // 16M elements
    __bf16* Qh  = (__bf16*)d_ws;
    __bf16* Ql  = Qh + qsz;
    __bf16* Mh  = Ql + qsz;
    __bf16* Ml  = Mh + msz;
    __bf16* MTh = Ml + msz;
    __bf16* MTl = MTh + msz;

    dpa_split_kernel<<<(unsigned)(qsz / (256 * 4)), 256, 0, stream>>>(Q, Qh, Ql);
    dpa_split_transpose_m_kernel<<<dim3(D_ / 32, TE_ / 32, N_), 256, 0, stream>>>(M, Mh, Ml, MTh, MTl);
    dpa_qk_kernel<<<dim3(TE_ / 128, TD_ / 128, N_), 256, 0, stream>>>(Qh, Ql, Mh, Ml, lengths, align);
    dpa_softmax_kernel<<<N_ * TD_, 256, 0, stream>>>(align);
    dpa_av_kernel<<<dim3(D_ / 128, TD_ / 128, N_), 256, 0, stream>>>(align, MTh, MTl, ctx);
}
